// SoftTreeEnsembleLayerTensor_14620068675646
// MI455X (gfx1250) — compile-verified, matrix operand order checked by reading, not run on hardware
//
#include <hip/hip_runtime.h>

// Problem constants (from reference): B=8192, F=512, T=64, D=6, O=128
// N_INTERNAL=63, N_LEAVES=64.  Node-GEMM columns = 63*64 = 4032.
// Leaf-GEMM K dimension = 64*64 = 4096.
// This version: M-tile = 32 rows per workgroup -> each B fragment fetched from
// L2 feeds TWO v_wmma instructions (rows 0-15 and 16-31), halving B traffic.

typedef __attribute__((ext_vector_type(16))) __bf16       v16bf;
typedef __attribute__((ext_vector_type(8)))  float        v8f;
typedef __attribute__((ext_vector_type(4)))  float        v4f;
typedef __attribute__((ext_vector_type(4)))  unsigned int v4u;
typedef __attribute__((ext_vector_type(2)))  unsigned int v2u;

static __device__ __forceinline__ unsigned short f2bf(float f) {
    union { float f; unsigned int u; } v; v.f = f;
    unsigned int u = v.u + 0x7FFFu + ((v.u >> 16) & 1u);   // round-to-nearest-even
    return (unsigned short)(u >> 16);
}
static __device__ __forceinline__ float bf2f(unsigned short h) {
    union { unsigned int u; float f; } v; v.u = ((unsigned int)h) << 16;
    return v.f;
}

union Frag { v4u q[2]; v16bf v; };

static __device__ __forceinline__ v16bf load_frag(const unsigned short* p0,
                                                  const unsigned short* p1) {
    Frag f;
    f.q[0] = *(const v4u*)p0;   // 16B chunk: K = off .. off+7     (pairs per VGPR)
    f.q[1] = *(const v4u*)p1;   // 16B chunk: K = 16+off .. 23+off
    return f.v;
}

// ---------- prep: W_nodes (63,512,64) f32 -> Wb[col=n*64+t][f] bf16 ----------
__global__ __launch_bounds__(256) void prep_wb(const float* __restrict__ Wn,
                                               unsigned short* __restrict__ wb) {
    int idx = blockIdx.x * 256 + threadIdx.x;          // 4032*512 total, exact
    int col = idx >> 9;
    int f   = idx & 511;
    int n   = col >> 6;
    int t   = col & 63;
    wb[idx] = f2bf(Wn[((size_t)n * 512 + f) * 64 + t]);
}

// ---------- prep: W_leaf (64,128,64) f32 -> WlT[o][k=l*64+t] bf16 ----------
__global__ __launch_bounds__(256) void prep_wl(const float* __restrict__ Wl,
                                               unsigned short* __restrict__ wlt) {
    int idx = blockIdx.x * 256 + threadIdx.x;          // 128*4096 total, exact
    int o   = idx >> 12;
    int k   = idx & 4095;
    int l   = k >> 6;
    int t   = k & 63;
    wlt[idx] = f2bf(Wl[((size_t)l * 128 + o) * 64 + t]);
}

// ---------- fused: node-GEMM + smooth_step + tree expansion + leaf-GEMM ----------
__global__ __launch_bounds__(256) void soft_tree_fused(
        const float* __restrict__ x,            // (8192, 512) f32
        const float* __restrict__ bn,           // (63, 64)    f32, flat col = n*64+t
        const unsigned short* __restrict__ wb,  // (4032, 512) bf16
        const unsigned short* __restrict__ wlt, // (128, 4096) bf16
        float* __restrict__ out) {              // (8192, 128) f32

    __shared__ __align__(16) unsigned short lds_x[32 * 512];   // 32 KB  (x tile, bf16)
    __shared__ __align__(16) unsigned short lds_t[32 * 4096];  // 256 KB (s, then prob)

    const int tid     = threadIdx.x;
    const int lane    = tid & 31;
    const int wid     = tid >> 5;               // 8 waves
    const int rowBase = blockIdx.x * 32;

    const int nhalf = lane >> 4;                // 0 / 1 (lane half)
    const int nl    = lane & 15;                // row (A) / column (B) index
    const int off   = nhalf * 8;                // K sub-offset per ISA 16-bit layout

    // -------- phase 1: stage x tile into LDS as bf16 (32 x 512) --------
#pragma unroll
    for (int i = 0; i < 16; ++i) {
        int idx4 = tid + i * 256;               // 4096 float4 groups
        int row  = idx4 >> 7;                   // 128 float4 per row
        int f4   = (idx4 & 127) * 4;
        v4f v = *(const v4f*)(x + (size_t)(rowBase + row) * 512 + f4);
        unsigned int p0 = (unsigned int)f2bf(v.x) | ((unsigned int)f2bf(v.y) << 16);
        unsigned int p1 = (unsigned int)f2bf(v.z) | ((unsigned int)f2bf(v.w) << 16);
        v2u pk; pk.x = p0; pk.y = p1;
        *(v2u*)(&lds_x[row * 512 + f4]) = pk;
    }
    __syncthreads();

    // -------- phase 2: node GEMM, 252 N-tiles of 16 cols, K=512, 2 M-tiles --------
    for (int nt = wid; nt < 252; nt += 8) {
        const int colBase = nt << 4;
        const float bias  = bn[colBase + nl];   // bias depends only on column
        v8f acc0, acc1;
#pragma unroll
        for (int r = 0; r < 8; ++r) { acc0[r] = bias; acc1[r] = bias; }

#pragma unroll
        for (int kk = 0; kk < 512; kk += 32) {
            const unsigned short* bp = wb + (size_t)(colBase + nl) * 512 + kk;
            v16bf b = load_frag(bp + off, bp + 16 + off);       // fetched ONCE
            const unsigned short* a0p = &lds_x[nl * 512 + kk];
            const unsigned short* a1p = &lds_x[(nl + 16) * 512 + kk];
            v16bf a0 = load_frag(a0p + off, a0p + 16 + off);    // rows 0..15
            v16bf a1 = load_frag(a1p + off, a1p + 16 + off);    // rows 16..31
            acc0 = __builtin_amdgcn_wmma_f32_16x16x32_bf16(
                       false, a0, false, b, (short)0, acc0, false, false);
            acc1 = __builtin_amdgcn_wmma_f32_16x16x32_bf16(
                       false, a1, false, b, (short)0, acc1, false, false);
        }
        // smooth_step + store s tile (bf16) into LDS for both M-tiles
#pragma unroll
        for (int r = 0; r < 8; ++r) {
            int m = r + nhalf * 8;              // C layout: lanes16-31 hold M+8
            float v0  = acc0[r];
            float tc0 = fminf(fmaxf(v0, -0.5f), 0.5f);
            lds_t[m * 4096 + colBase + nl] =
                f2bf(tc0 * (1.5f - 2.0f * tc0 * tc0) + 0.5f);
            float v1  = acc1[r];
            float tc1 = fminf(fmaxf(v1, -0.5f), 0.5f);
            lds_t[(m + 16) * 4096 + colBase + nl] =
                f2bf(tc1 * (1.5f - 2.0f * tc1 * tc1) + 0.5f);
        }
    }
    __syncthreads();

    // -------- phase 3: tree path-probability expansion (in place) --------
    // Each (row, t) column is private to one thread-iteration: no barriers needed.
#pragma unroll 1
    for (int i = 0; i < 8; ++i) {
        int cidx = tid + i * 256;               // 2048 (row, t) columns
        int row  = cidx >> 6;
        int tcol = cidx & 63;
        const int base = row * 4096 + tcol;

        float p[64];
        p[0] = 1.0f;
#pragma unroll
        for (int lvl = 0; lvl < 6; ++lvl) {
            const int width = 1 << lvl;
            const int nbase = width - 1;        // first node index of this level
#pragma unroll
            for (int j = width - 1; j >= 0; --j) {   // descending: in-place safe
                float s  = bf2f(lds_t[base + (nbase + j) * 64]);
                float pj = p[j];
                p[2 * j + 1] = pj * (1.0f - s);
                p[2 * j]     = pj * s;
            }
        }
        // overwrite s region with leaf probabilities (same column only)
#pragma unroll
        for (int l = 0; l < 64; ++l)
            lds_t[base + l * 64] = f2bf(p[l]);
    }
    __syncthreads();

    // -------- phase 4: leaf GEMM, out(32x128) = prob(32x4096) @ WlT^T --------
    const int oc = wid * 16 + nl;               // output column, one N-tile per wave
    v8f acc0, acc1;
#pragma unroll
    for (int r = 0; r < 8; ++r) { acc0[r] = 0.0f; acc1[r] = 0.0f; }

#pragma unroll 8
    for (int kk = 0; kk < 4096; kk += 32) {
        const unsigned short* bp = wlt + (size_t)oc * 4096 + kk;
        v16bf b = load_frag(bp + off, bp + 16 + off);           // fetched ONCE
        const unsigned short* a0p = &lds_t[nl * 4096 + kk];
        const unsigned short* a1p = &lds_t[(nl + 16) * 4096 + kk];
        v16bf a0 = load_frag(a0p + off, a0p + 16 + off);
        v16bf a1 = load_frag(a1p + off, a1p + 16 + off);
        acc0 = __builtin_amdgcn_wmma_f32_16x16x32_bf16(
                   false, a0, false, b, (short)0, acc0, false, false);
        acc1 = __builtin_amdgcn_wmma_f32_16x16x32_bf16(
                   false, a1, false, b, (short)0, acc1, false, false);
    }
#pragma unroll
    for (int r = 0; r < 8; ++r) {
        int m = r + nhalf * 8;
        out[(size_t)(rowBase + m) * 128 + oc]      = acc0[r];
        out[(size_t)(rowBase + 16 + m) * 128 + oc] = acc1[r];
    }
}

extern "C" void kernel_launch(void* const* d_in, const int* in_sizes, int n_in,
                              void* d_out, int out_size, void* d_ws, size_t ws_size,
                              hipStream_t stream) {
    const float* x  = (const float*)d_in[0];   // (8192, 512)
    const float* Wn = (const float*)d_in[1];   // (63, 512, 64)
    const float* bn = (const float*)d_in[2];   // (63, 64)
    const float* Wl = (const float*)d_in[3];   // (64, 128, 64)
    float* out = (float*)d_out;                // (8192, 128)

    unsigned short* wb  = (unsigned short*)d_ws;                       // 4032*512*2 B
    unsigned short* wlt = (unsigned short*)((char*)d_ws + (size_t)4032 * 512 * 2);

    prep_wb<<<(4032 * 512) / 256, 256, 0, stream>>>(Wn, wb);
    prep_wl<<<(128 * 4096) / 256, 256, 0, stream>>>(Wl, wlt);
    soft_tree_fused<<<8192 / 32, 256, 0, stream>>>(x, bn, wb, wlt, out);
}
